// MambaBlock_86595130622189
// MI455X (gfx1250) — compile-verified, matrix operand order checked by reading, not run on hardware
//
#include <hip/hip_runtime.h>
#include <hip/hip_bf16.h>
#include <math.h>

// ---------------------------------------------------------------------------
// Mamba block for MI455X (gfx1250, wave32, WMMA, bf16 operands / f32 accum).
// B=2, L=1024, D_MODEL=1024, D_INNER=2048, D_STATE=16, DT_RANK=64, D_CONV=4.
// ---------------------------------------------------------------------------

#define BB       2
#define LL       1024
#define DMODEL   1024
#define DINNER   2048
#define DSTATE   16
#define DTRANK   64
#define MM       (BB * LL)               // 2048 token rows
#define XDW      (DTRANK + 2 * DSTATE)   // 96

typedef __attribute__((ext_vector_type(16))) __bf16 v16bf;
typedef __attribute__((ext_vector_type(8)))  __bf16 v8bf;
typedef __attribute__((ext_vector_type(4)))  __bf16 v4bf;
typedef __attribute__((ext_vector_type(8)))  float  v8f;

// Fragment: two 16-byte chunks of 8 bf16 (ISA 16-bit A/B layout: elems 0..7 =
// K octet at p0, elems 8..15 = K octet at p1).  Works on global or LDS ptrs.
__device__ __forceinline__ v16bf fragbf(const __bf16* __restrict__ p0,
                                        const __bf16* __restrict__ p1) {
  const v8bf lo = *(const v8bf*)p0;
  const v8bf hi = *(const v8bf*)p1;
  return __builtin_shufflevector(lo, hi, 0, 1, 2, 3, 4, 5, 6, 7,
                                         8, 9, 10, 11, 12, 13, 14, 15);
}

__device__ __forceinline__ float softplus_f(float v) {
  return (v > 20.f) ? v : log1pf(__expf(v));
}
__device__ __forceinline__ float silu_f(float v) {
  return v / (1.f + __expf(-v));
}

// ---------------------------------------------------------------------------
// Elementwise f32 -> bf16 (vectorized x4; element counts all divisible by 4).
// ---------------------------------------------------------------------------
__global__ void cvt_bf16x4_kernel(const float* __restrict__ s,
                                  __bf16* __restrict__ d, int n4) {
  const int i = blockIdx.x * blockDim.x + threadIdx.x;
  if (i >= n4) return;
  const float4 v = ((const float4*)s)[i];
  v4bf o;
  o[0] = (__bf16)v.x; o[1] = (__bf16)v.y; o[2] = (__bf16)v.z; o[3] = (__bf16)v.w;
  ((v4bf*)d)[i] = o;
}

// ---------------------------------------------------------------------------
// LDS-tiled GEMM: Y[M x N](f32) = X[M x K](bf16) * W[N x K](bf16)^T.
// Block = 256 threads = 8 waves (4 M x 2 N); block tile 128x128; K-step 32.
// Double-buffered LDS panels (row stride padded to 40 bf16 = 80 B so the
// 16-lane ds_load_b128 fragment reads hit distinct bank groups).
// Per wave: 32x64 strip = 2 M-frags x 4 N-frags = 8 WMMA / K-step, B frags
// reused across both M-frags.  ~64 FLOP per L2 byte (vs ~22 direct).
// ACT: 0 = none, 1 = bias + softplus.  Requires M%128==0, N%128==0, K%32==0.
// ---------------------------------------------------------------------------
#define BM   128
#define BN   128
#define KS   32
#define PAD  40            // padded LDS row stride in bf16 elements (80 B)

template <int ACT>
__global__ __launch_bounds__(256) void gemm_lds_bf16_wmma(
    const __bf16* __restrict__ X, int ldx,
    const __bf16* __restrict__ W, int ldw,
    float* __restrict__ Y, int ldy,
    int M, int N, int K, const float* __restrict__ bias)
{
  __shared__ __bf16 As[2][BM * PAD];
  __shared__ __bf16 Bs[2][BN * PAD];

  const int tid  = threadIdx.x;
  const int lane = tid & 31;
  const int wave = tid >> 5;
  const int half = lane >> 4;      // 0: K octets {0,16}; 1: K octets {8,24}
  const int lm   = lane & 15;
  const int wm   = wave >> 1;      // 0..3  (M position of wave)
  const int wn   = wave & 1;       // 0..1  (N position of wave)

  const int Nb = N / BN;
  const int mb = blockIdx.x / Nb;
  const int nb = blockIdx.x % Nb;

  const __bf16* __restrict__ Ag = X + (size_t)(mb * BM) * ldx;
  const __bf16* __restrict__ Bg = W + (size_t)(nb * BN) * ldw;

  // Cooperative staging: each panel is 128 rows x 4 chunks of 16 B;
  // thread moves rows r0 and r0+64, chunk c0, for both panels.
  const int r0 = tid >> 2;         // 0..63
  const int c0 = tid & 3;          // chunk-in-row (16 B units)

  auto stage = [&](int buf, int k0) {
    const v8bf a0 = *(const v8bf*)(Ag + (size_t)r0        * ldx + k0 + c0 * 8);
    const v8bf a1 = *(const v8bf*)(Ag + (size_t)(r0 + 64) * ldx + k0 + c0 * 8);
    const v8bf b0 = *(const v8bf*)(Bg + (size_t)r0        * ldw + k0 + c0 * 8);
    const v8bf b1 = *(const v8bf*)(Bg + (size_t)(r0 + 64) * ldw + k0 + c0 * 8);
    *(v8bf*)(&As[buf][r0        * PAD + c0 * 8]) = a0;
    *(v8bf*)(&As[buf][(r0 + 64) * PAD + c0 * 8]) = a1;
    *(v8bf*)(&Bs[buf][r0        * PAD + c0 * 8]) = b0;
    *(v8bf*)(&Bs[buf][(r0 + 64) * PAD + c0 * 8]) = b1;
  };

  v8f acc0[4], acc1[4];
#pragma unroll
  for (int i = 0; i < 4; ++i) { acc0[i] = (v8f){}; acc1[i] = (v8f){}; }

  stage(0, 0);
  __syncthreads();

  const int nk = K / KS;
  for (int ks = 0; ks < nk; ++ks) {
    const int cur = ks & 1;
    if (ks + 1 < nk) stage(cur ^ 1, (ks + 1) * KS);   // overlap with WMMA

    const __bf16* a0p = &As[cur][(wm * 32 + lm) * PAD + half * 8];
    const __bf16* a1p = a0p + 16 * PAD;
    const v16bf a0 = fragbf(a0p, a0p + 16);
    const v16bf a1 = fragbf(a1p, a1p + 16);
#pragma unroll
    for (int nt = 0; nt < 4; ++nt) {
      const __bf16* bp = &Bs[cur][(wn * 64 + nt * 16 + lm) * PAD + half * 8];
      const v16bf b = fragbf(bp, bp + 16);
      acc0[nt] = __builtin_amdgcn_wmma_f32_16x16x32_bf16(false, a0, false, b,
                                                         (short)0, acc0[nt],
                                                         false, false);
      acc1[nt] = __builtin_amdgcn_wmma_f32_16x16x32_bf16(false, a1, false, b,
                                                         (short)0, acc1[nt],
                                                         false, false);
    }
    __syncthreads();
  }

  // C/D layout: lane<16 -> col=lane, rows +0..7; lane>=16 -> rows +8..15.
  const int rb0 = mb * BM + wm * 32 + half * 8;
  const int rb1 = rb0 + 16;
#pragma unroll
  for (int nt = 0; nt < 4; ++nt) {
    const int col = nb * BN + wn * 64 + nt * 16 + lm;
    const float bv = (ACT == 1) ? bias[col] : 0.f;
#pragma unroll
    for (int j = 0; j < 8; ++j) {
      float v0 = acc0[nt][j] + bv;
      float v1 = acc1[nt][j] + bv;
      if (ACT == 1) { v0 = softplus_f(v0); v1 = softplus_f(v1); }
      Y[(size_t)(rb0 + j) * ldy + col] = v0;
      Y[(size_t)(rb1 + j) * ldy + col] = v1;
    }
  }
}

// ---------------------------------------------------------------------------
// Direct-global GEMM for the skinny x_proj (N=96: NT=3 -> 48-wide groups).
// One wave -> 32 x 48 strip, branch-free K-loop.
// ---------------------------------------------------------------------------
template <int NT>
__global__ void gemm_bf16_wmma(const __bf16* __restrict__ X, int ldx,
                               const __bf16* __restrict__ W, int ldw,
                               float* __restrict__ Y, int ldy,
                               int M, int N, int K)
{
  const int lane = threadIdx.x & 31;
  const int wave = threadIdx.x >> 5;
  const int half = lane >> 4;
  const int lm   = lane & 15;

  const int NG   = N / (NT * 16);
  const int jobs = (M >> 5) * NG;
  const int job  = blockIdx.x * 8 + wave;
  if (job >= jobs) return;
  const int mt = job / NG;
  const int ng = job % NG;
  const int nb = ng * (NT * 16);

  const __bf16* __restrict__ xr0 = X + (size_t)(mt * 32 + lm) * ldx;
  const __bf16* __restrict__ xr1 = xr0 + (size_t)16 * ldx;

  v8f acc0[NT], acc1[NT];
#pragma unroll
  for (int i = 0; i < NT; ++i) { acc0[i] = (v8f){}; acc1[i] = (v8f){}; }

  for (int k0 = 0; k0 < K; k0 += 32) {
    const int ka = k0 + half * 8;
    const int kb = ka + 16;
    if (k0 + 32 < K) {
      __builtin_prefetch(xr0 + k0 + 32, 0, 1);
      __builtin_prefetch(xr1 + k0 + 32, 0, 1);
    }
    const v16bf a0 = fragbf(xr0 + ka, xr0 + kb);
    const v16bf a1 = fragbf(xr1 + ka, xr1 + kb);
#pragma unroll
    for (int nt = 0; nt < NT; ++nt) {
      const __bf16* __restrict__ wp = W + (size_t)(nb + nt * 16 + lm) * ldw;
      const v16bf b = fragbf(wp + ka, wp + kb);
      acc0[nt] = __builtin_amdgcn_wmma_f32_16x16x32_bf16(false, a0, false, b,
                                                         (short)0, acc0[nt],
                                                         false, false);
      acc1[nt] = __builtin_amdgcn_wmma_f32_16x16x32_bf16(false, a1, false, b,
                                                         (short)0, acc1[nt],
                                                         false, false);
    }
  }

  const int rb0 = mt * 32 + half * 8;
  const int rb1 = rb0 + 16;
#pragma unroll
  for (int nt = 0; nt < NT; ++nt) {
    const int col = nb + nt * 16 + lm;
#pragma unroll
    for (int j = 0; j < 8; ++j) {
      Y[(size_t)(rb0 + j) * ldy + col] = acc0[nt][j];
      Y[(size_t)(rb1 + j) * ldy + col] = acc1[nt][j];
    }
  }
}

// ---------------------------------------------------------------------------
// Depthwise causal conv (k=4) + bias + SiLU.  xz = first DINNER columns of Y1
// (row stride 2*DINNER).  Writes u as f32 (for the scan) and bf16 (for GEMMs).
// ---------------------------------------------------------------------------
__global__ void conv_silu_kernel(const float* __restrict__ Y1,
                                 const float* __restrict__ Wc,
                                 const float* __restrict__ bc,
                                 float* __restrict__ U,
                                 __bf16* __restrict__ Ubf)
{
  const int idx = blockIdx.x * blockDim.x + threadIdx.x;   // b*L*D + t*D + d
  if (idx >= BB * LL * DINNER) return;
  const int d = idx & (DINNER - 1);
  const int t = (idx >> 11) & (LL - 1);
  const int b = idx >> 21;
  const float* __restrict__ xz = Y1 + (size_t)b * LL * (2 * DINNER);
  float acc = bc[d];
#pragma unroll
  for (int j = 0; j < 4; ++j) {
    const int tt = t - 3 + j;
    if (tt >= 0) acc += Wc[d * 4 + j] * xz[(size_t)tt * (2 * DINNER) + d];
  }
  const float s = silu_f(acc);
  U[idx]   = s;
  Ubf[idx] = (__bf16)s;
}

// ---------------------------------------------------------------------------
// Selective scan.  Each (b,d) channel -> 16 lanes (lane = state index n),
// wave32 -> 2 channels/wave.  h stays in a register; h*C reduced across the
// 16 states with a shfl_xor butterfly.  Fuses +u*D and the res gate; stores
// gated y directly as bf16 for out_proj.
// ---------------------------------------------------------------------------
__global__ void scan_kernel(const float* __restrict__ DELTA,
                            const float* __restrict__ U,
                            const float* __restrict__ XDBL,
                            const float* __restrict__ A_log,
                            const float* __restrict__ Dvec,
                            const float* __restrict__ Y1,
                            __bf16* __restrict__ Ygbf)
{
  const int lane = threadIdx.x & 31;
  const int wave = threadIdx.x >> 5;
  const int half = lane >> 4;
  const int n    = lane & 15;
  const int ch   = blockIdx.x * 16 + wave * 2 + half;      // 0..BB*DINNER-1
  const int b    = ch >> 11;
  const int d    = ch & (DINNER - 1);

  const float a  = -__expf(A_log[d * DSTATE + n]);
  const float Dd = Dvec[d];

  const size_t baseU = (size_t)b * LL * DINNER + d;
  const size_t baseX = (size_t)b * LL * XDW;
  const size_t baseR = (size_t)b * LL * (2 * DINNER) + DINNER + d;

  float h = 0.f;
  for (int t = 0; t < LL; ++t) {
    const float dt = DELTA[baseU + (size_t)t * DINNER];
    const float ut = U[baseU + (size_t)t * DINNER];
    const float Bn = XDBL[baseX + t * XDW + DTRANK + n];
    const float Cn = XDBL[baseX + t * XDW + DTRANK + DSTATE + n];

    h = __expf(dt * a) * h + (dt * ut) * Bn;

    float p = h * Cn;                       // reduce over the 16 states
    p += __shfl_xor(p, 1, 32);
    p += __shfl_xor(p, 2, 32);
    p += __shfl_xor(p, 4, 32);
    p += __shfl_xor(p, 8, 32);

    const float r  = Y1[baseR + (size_t)t * (2 * DINNER)];
    const float yg = (p + ut * Dd) * silu_f(r);
    if (n == 0) Ygbf[baseU + (size_t)t * DINNER] = (__bf16)yg;
  }
}

// ---------------------------------------------------------------------------
extern "C" void kernel_launch(void* const* d_in, const int* in_sizes, int n_in,
                              void* d_out, int out_size, void* d_ws, size_t ws_size,
                              hipStream_t stream)
{
  const float* x       = (const float*)d_in[0];   // (B,L,D_MODEL)
  const float* W_in    = (const float*)d_in[1];   // (2*DINNER, DMODEL)
  const float* W_conv  = (const float*)d_in[2];   // (DINNER,1,4)
  const float* b_conv  = (const float*)d_in[3];   // (DINNER)
  const float* W_xproj = (const float*)d_in[4];   // (XDW, DINNER)
  const float* W_dt    = (const float*)d_in[5];   // (DINNER, DTRANK)
  const float* b_dt    = (const float*)d_in[6];   // (DINNER)
  const float* A_log   = (const float*)d_in[7];   // (DINNER, DSTATE)
  const float* Dv      = (const float*)d_in[8];   // (DINNER)
  const float* W_out   = (const float*)d_in[9];   // (DMODEL, DINNER)
  float* out = (float*)d_out;                     // (B,L,DMODEL)

  // ---- workspace carve-up ----
  char* p = (char*)d_ws;
  float* Y1 = (float*)p;  p += (size_t)MM * (2 * DINNER) * 4;   // xz | res
  float* U  = (float*)p;  p += (size_t)MM * DINNER * 4;         // u (f32)
  float* XD = (float*)p;  p += (size_t)MM * XDW * 4;            // x_dbl (f32)
  float* DE = (float*)p;  p += (size_t)MM * DINNER * 4;         // delta
  __bf16* Xbf    = (__bf16*)p; p += (size_t)MM * DMODEL * 2;
  __bf16* Winbf  = (__bf16*)p; p += (size_t)(2 * DINNER) * DMODEL * 2;
  __bf16* Wxbf   = (__bf16*)p; p += (size_t)XDW * DINNER * 2;
  __bf16* Wdtbf  = (__bf16*)p; p += (size_t)DINNER * DTRANK * 2;
  __bf16* Woutbf = (__bf16*)p; p += (size_t)DMODEL * DINNER * 2;
  __bf16* Ubf    = (__bf16*)p; p += (size_t)MM * DINNER * 2;
  __bf16* XDbf   = (__bf16*)p; p += (size_t)MM * XDW * 2;
  __bf16* Ygbf   = (__bf16*)p; p += (size_t)MM * DINNER * 2;

  // ---- 0. one-shot f32 -> bf16 of x and all weight matrices ----
  cvt_bf16x4_kernel<<<(MM * DMODEL / 4) / 256, 256, 0, stream>>>(x, Xbf, MM * DMODEL / 4);
  cvt_bf16x4_kernel<<<(2 * DINNER * DMODEL / 4) / 256, 256, 0, stream>>>(W_in, Winbf, 2 * DINNER * DMODEL / 4);
  cvt_bf16x4_kernel<<<(XDW * DINNER / 4) / 256, 256, 0, stream>>>(W_xproj, Wxbf, XDW * DINNER / 4);
  cvt_bf16x4_kernel<<<(DINNER * DTRANK / 4) / 256, 256, 0, stream>>>(W_dt, Wdtbf, DINNER * DTRANK / 4);
  cvt_bf16x4_kernel<<<(DMODEL * DINNER / 4) / 256, 256, 0, stream>>>(W_out, Woutbf, DMODEL * DINNER / 4);

  // ---- 1. in_proj: Y1 = x @ W_in^T  (2048 x 4096, K=1024) ----
  gemm_lds_bf16_wmma<0><<<(MM / BM) * (2 * DINNER / BN), 256, 0, stream>>>(
      Xbf, DMODEL, Winbf, DMODEL, Y1, 2 * DINNER, MM, 2 * DINNER, DMODEL, nullptr);

  // ---- 2. depthwise causal conv + SiLU -> U (f32 + bf16) ----
  conv_silu_kernel<<<(BB * LL * DINNER) / 256, 256, 0, stream>>>(Y1, W_conv, b_conv, U, Ubf);

  // ---- 3. x_dbl = u @ W_xproj^T  (2048 x 96, K=2048; NT=3 -> 48-wide) ----
  gemm_bf16_wmma<3><<<(MM / 32) * (XDW / 48) / 8, 256, 0, stream>>>(
      Ubf, DINNER, Wxbf, DINNER, XD, XDW, MM, XDW, DINNER);

  // ---- 3b. bf16 copy of x_dbl (dt columns feed GEMM 4 as A operand) ----
  cvt_bf16x4_kernel<<<(MM * XDW / 4) / 256, 256, 0, stream>>>(XD, XDbf, MM * XDW / 4);

  // ---- 4. delta = softplus(dt @ W_dt^T + b_dt)  (2048 x 2048, K=64) ----
  gemm_lds_bf16_wmma<1><<<(MM / BM) * (DINNER / BN), 256, 0, stream>>>(
      XDbf, XDW, Wdtbf, DTRANK, DE, DINNER, MM, DINNER, DTRANK, b_dt);

  // ---- 5. selective scan + D skip + res gate -> Ygbf (bf16) ----
  scan_kernel<<<(BB * DINNER) / 16, 256, 0, stream>>>(DE, U, XD, A_log, Dv, Y1, Ygbf);

  // ---- 6. out_proj: out = yg @ W_out^T  (2048 x 1024, K=2048) ----
  gemm_lds_bf16_wmma<0><<<(MM / BM) * (DMODEL / BN), 256, 0, stream>>>(
      Ygbf, DINNER, Woutbf, DINNER, out, DMODEL, MM, DMODEL, DINNER, nullptr);
}
